// FalconH1Mixer_11089605558909
// MI455X (gfx1250) — matrix operand order repaired and checked by reading, not compiled, not run on hardware
//
#include <hip/hip_runtime.h>
#include <hip/hip_bf16.h>
#include <stdint.h>

// ---------------- problem constants ----------------
#define B_SZ     2
#define L_SZ     4096
#define HDIM     2048
#define D_INNER  4096
#define N_HEADS  64
#define HEAD_D   64
#define D_STATE  128
#define CONV_K   4
#define CHUNK    128
#define NCHUNK   (L_SZ / CHUNK)                    // 32
#define CONV_DIM (D_INNER + 2 * D_STATE)           // 4352
#define PROJ_DIM (D_INNER + CONV_DIM + N_HEADS)    // 8512
#define TOK      (B_SZ * L_SZ)                     // 8192
#define EPSF     1e-5f

// gfx1250 async global->LDS staging (ASYNCcnt path). Flip to 0 if assembler rejects.
#define ASYNC_COPY 1

typedef __attribute__((ext_vector_type(16))) __bf16 v16bf;
typedef __attribute__((ext_vector_type(8)))  __bf16 v8bf;
typedef __attribute__((ext_vector_type(8)))  float  v8f;

__device__ inline __bf16 bfc(float x) { return (__bf16)x; }
__device__ inline float  siluf(float x) { return x / (1.f + __expf(-x)); }

__device__ inline v8f wmma_bf16(v16bf a, v16bf b, v8f c) {
    return __builtin_amdgcn_wmma_f32_16x16x32_bf16(false, a, false, b, (short)0, c,
                                                   false, false);
}

// A operand (16x32 bf16): lane m = lane&15; 16 contiguous K at 16*(lane>>4).
__device__ inline v16bf load_a(const __bf16* base, int stride, int row0, int kbase, int lane) {
    const int m  = lane & 15;
    const int kh = lane >> 4;
    const __bf16* p = base + (size_t)(row0 + m) * stride + kbase + kh * 16;
    v8bf lo = *(const v8bf*)p;
    v8bf hi = *(const v8bf*)(p + 8);
    return __builtin_shufflevector(lo, hi, 0, 1, 2, 3, 4, 5, 6, 7,
                                   8, 9, 10, 11, 12, 13, 14, 15);
}

// B operand (32x16 bf16): LDS holds Bmat as [col][k] row-major; lane col = lane&15,
// 16 contiguous K halves at 16*(lane>>4).
__device__ inline v16bf load_b(const __bf16* base, int stride, int col0, int kbase, int lane) {
    const int n  = lane & 15;
    const int kh = lane >> 4;
    const __bf16* p = base + (size_t)(col0 + n) * stride + kbase + kh * 16;
    return *(const v16bf*)p;
}

// 16-byte global -> LDS copy, async (ASYNCcnt) when enabled.
__device__ inline void cp16(__bf16* lds, const __bf16* g) {
#if ASYNC_COPY
    asm volatile("global_load_async_to_lds_b128 %0, %1, off"
                 :: "v"((uint32_t)(uintptr_t)lds), "v"(g) : "memory");
#else
    *(v8bf*)lds = *(const v8bf*)g;
#endif
}

// ================= 0) f32 -> bf16 precision-convert pass =================
__global__ void k_f2bf(const float* __restrict__ in, __bf16* __restrict__ out, size_t n) {
    size_t i = ((size_t)blockIdx.x * blockDim.x + threadIdx.x) * 4;
    if (i >= n) return;
    float4 v = *(const float4*)(in + i);
    out[i + 0] = bfc(v.x); out[i + 1] = bfc(v.y);
    out[i + 2] = bfc(v.z); out[i + 3] = bfc(v.w);
}

// ================= 1) bf16 GEMM: out[M][N] = A[M][K] @ W[N][K]^T ================
// BM=256, BN=64, BK=64; 8 waves as 4(M) x 2(N), wave tile 64x32.
// Double-buffered LDS staging via async global->LDS (10 x b128 per thread per tile).
#define BM 256
#define BN 64
#define BK 64
__global__ __launch_bounds__(256) void k_gemm_bf16(const __bf16* __restrict__ A,
                                                   const __bf16* __restrict__ Bw,
                                                   float* __restrict__ out,
                                                   int M, int N, int K) {
    __shared__ __bf16 sA[2][BM * BK];
    __shared__ __bf16 sB[2][BN * BK];
    const int mBase = blockIdx.y * BM;
    const int nBase = blockIdx.x * BN;
    const int tid = threadIdx.x, lane = tid & 31, wave = tid >> 5;
    const int wm = (wave >> 1) * 64;   // 4 wave rows of 64
    const int wn = (wave & 1) * 32;    // 2 wave cols of 32
    v8f acc[4][2] = {};

    auto issueTile = [&](int buf, int k0) {
        // A tile BM x BK: (BM*BK)/8 = 2048 16B chunks -> 8 per thread
#pragma unroll
        for (int i = 0; i < (BM * BK) / (256 * 8); ++i) {
            int o = tid + i * 256;
            int row = o >> 3, ck = (o & 7) * 8;
            cp16(&sA[buf][row * BK + ck], A + (size_t)(mBase + row) * K + k0 + ck);
        }
        // B tile BN x BK: 512 chunks -> 2 per thread
#pragma unroll
        for (int i = 0; i < (BN * BK) / (256 * 8); ++i) {
            int o = tid + i * 256;
            int row = o >> 3, ck = (o & 7) * 8;
            cp16(&sB[buf][row * BK + ck], Bw + (size_t)(nBase + row) * K + k0 + ck);
        }
    };

    issueTile(0, 0);
    int buf = 0;
    for (int k0 = 0; k0 < K; k0 += BK) {
        const bool more = (k0 + BK) < K;
        if (more) issueTile(buf ^ 1, k0 + BK);
#if ASYNC_COPY
        if (more) asm volatile("s_wait_asynccnt 0xa" ::: "memory");  // cur tile done, next in flight
        else      asm volatile("s_wait_asynccnt 0x0" ::: "memory");
#endif
        __syncthreads();
#pragma unroll
        for (int kk = 0; kk < BK; kk += 32) {
            v16bf b0 = load_b(sB[buf], BK, wn, kk, lane);
            v16bf b1 = load_b(sB[buf], BK, wn + 16, kk, lane);
#pragma unroll
            for (int mi = 0; mi < 4; ++mi) {
                v16bf a = load_a(sA[buf], BK, wm + mi * 16, kk, lane);
                acc[mi][0] = wmma_bf16(a, b0, acc[mi][0]);
                acc[mi][1] = wmma_bf16(a, b1, acc[mi][1]);
            }
        }
        __syncthreads();
        buf ^= 1;
    }
    const int n = lane & 15, mh = (lane >> 4) * 8;
#pragma unroll
    for (int mi = 0; mi < 4; ++mi)
#pragma unroll
        for (int ni = 0; ni < 2; ++ni) {
            float* o = out + (size_t)(mBase + wm + mi * 16 + mh) * N
                           + nBase + wn + ni * 16 + n;
#pragma unroll
            for (int r = 0; r < 8; ++r) o[(size_t)r * N] = acc[mi][ni][r];
        }
}

// ================= 2) depthwise causal conv (K=4) + bias + SiLU
__global__ void k_conv(const float* __restrict__ proj, const float* __restrict__ cw,
                       const float* __restrict__ cb, float* __restrict__ xbc) {
    size_t idx = (size_t)blockIdx.x * blockDim.x + threadIdx.x;
    if (idx >= (size_t)TOK * CONV_DIM) return;
    int ch = (int)(idx % CONV_DIM);
    size_t t = idx / CONV_DIM;
    int l = (int)(t % L_SZ);
    size_t bBase = t - l;
    float acc = cb[ch];
#pragma unroll
    for (int k = 0; k < CONV_K; ++k) {
        int ll = l + k - (CONV_K - 1);
        if (ll >= 0)
            acc += proj[(bBase + ll) * (size_t)PROJ_DIM + D_INNER + ch] * cw[ch * CONV_K + k];
    }
    xbc[idx] = siluf(acc);
}

// ================= 3) dt = softplus(proj_dt + dt_bias)
__global__ void k_dt(const float* __restrict__ proj, const float* __restrict__ dt_bias,
                     float* __restrict__ dt) {
    int idx = blockIdx.x * blockDim.x + threadIdx.x;
    if (idx >= TOK * N_HEADS) return;
    int h = idx & (N_HEADS - 1);
    size_t t = (size_t)(idx >> 6);
    float z = proj[t * PROJ_DIM + D_INNER + CONV_DIM + h] + dt_bias[h];
    dt[idx] = (z > 20.f) ? z : log1pf(__expf(z));
}

// ================= 4) per-chunk inclusive cumsum of A*dt -> acs[b][h][c][t]
__global__ void k_acs(const float* __restrict__ dt, const float* __restrict__ A_log,
                      float* __restrict__ acs) {
    int idx = blockIdx.x * blockDim.x + threadIdx.x;   // (b*H + h)*NCHUNK + c
    if (idx >= B_SZ * N_HEADS * NCHUNK) return;
    int c = idx % NCHUNK;
    int h = (idx / NCHUNK) % N_HEADS;
    int b = idx / (NCHUNK * N_HEADS);
    float Aconst = -__expf(A_log[h]);
    float s = 0.f;
    float* o = acs + (size_t)idx * CHUNK;
    const float* d = dt + ((size_t)b * L_SZ + (size_t)c * CHUNK) * N_HEADS + h;
    for (int t = 0; t < CHUNK; ++t) { s += Aconst * d[(size_t)t * N_HEADS]; o[t] = s; }
}

// ================= 5) intra-chunk SSD: G, Y_diag, chunk states (all WMMA bf16)
__global__ __launch_bounds__(256) void k_chunk(const float* __restrict__ xbc,
                                               const float* __restrict__ dt,
                                               const float* __restrict__ acs,
                                               float* __restrict__ states,
                                               float* __restrict__ yacc) {
    const int h = blockIdx.x, c = blockIdx.y, b = blockIdx.z;
    __shared__ __bf16 sB [CHUNK * D_STATE];   // [s][n]
    __shared__ __bf16 sBT[D_STATE * CHUNK];   // [n][s], decay-scaled
    __shared__ __bf16 sC [CHUNK * D_STATE];   // [t][n]
    __shared__ __bf16 sXT[HEAD_D * CHUNK];    // [d][s]  (x * dt, transposed)
    __shared__ __bf16 sGL[CHUNK * CHUNK];     // [t][s]  (G * L)
    __shared__ float  sAcs[CHUNK];
    const int tid = threadIdx.x, lane = tid & 31, wave = tid >> 5;
    const size_t tokBase = (size_t)b * L_SZ + (size_t)c * CHUNK;
    if (tid < CHUNK)
        sAcs[tid] = acs[(((size_t)b * N_HEADS + h) * NCHUNK + c) * CHUNK + tid];
    {   // B and C chunk tiles (group-shared across heads)
        int row = tid >> 1, cb = (tid & 1) * 64;
        const float* gB = xbc + (tokBase + row) * CONV_DIM + D_INNER + cb;
        const float* gC = xbc + (tokBase + row) * CONV_DIM + D_INNER + D_STATE + cb;
#pragma unroll
        for (int i = 0; i < 64; i += 4) {
            float4 v = *(const float4*)(gB + i);
            sB[row * D_STATE + cb + i + 0] = bfc(v.x);
            sB[row * D_STATE + cb + i + 1] = bfc(v.y);
            sB[row * D_STATE + cb + i + 2] = bfc(v.z);
            sB[row * D_STATE + cb + i + 3] = bfc(v.w);
            float4 w = *(const float4*)(gC + i);
            sC[row * D_STATE + cb + i + 0] = bfc(w.x);
            sC[row * D_STATE + cb + i + 1] = bfc(w.y);
            sC[row * D_STATE + cb + i + 2] = bfc(w.z);
            sC[row * D_STATE + cb + i + 3] = bfc(w.w);
        }
    }
    {   // x*dt, stored transposed [d][s]
        int row = tid >> 1, cb = (tid & 1) * 32;    // row = t
        const float* gx = xbc + (tokBase + row) * CONV_DIM + (size_t)h * HEAD_D + cb;
        float dtv = dt[(tokBase + row) * N_HEADS + h];
#pragma unroll
        for (int i = 0; i < 32; i += 4) {
            float4 v = *(const float4*)(gx + i);
            sXT[(size_t)(cb + i + 0) * CHUNK + row] = bfc(v.x * dtv);
            sXT[(size_t)(cb + i + 1) * CHUNK + row] = bfc(v.y * dtv);
            sXT[(size_t)(cb + i + 2) * CHUNK + row] = bfc(v.z * dtv);
            sXT[(size_t)(cb + i + 3) * CHUNK + row] = bfc(v.w * dtv);
        }
    }
    __syncthreads();

    // ---- Phase 1: G[t][s] = sum_n C[t][n] B[s][n]; GL = G * exp(acs[t]-acs[s]) (causal)
    {
        const int tm = wave * 16;
        for (int s8 = 0; s8 < 8; ++s8) {
            v8f acc = {};
#pragma unroll
            for (int ks = 0; ks < 4; ++ks) {
                v16bf a = load_a(sC, D_STATE, tm, ks * 32, lane);
                v16bf bb = load_b(sB, D_STATE, s8 * 16, ks * 32, lane);
                acc = wmma_bf16(a, bb, acc);
            }
            const int nl = lane & 15, mh = (lane >> 4) * 8;
            const int s = s8 * 16 + nl;
#pragma unroll
            for (int r = 0; r < 8; ++r) {
                int t = tm + mh + r;
                float v = (t >= s) ? acc[r] * __expf(sAcs[t] - sAcs[s]) : 0.f;
                sGL[(size_t)t * CHUNK + s] = bfc(v);
            }
        }
    }
    __syncthreads();
    {   // decay-scaled transposed B: sBT[n][s] = B[s][n]*exp(acs[last]-acs[s])
        int row = tid >> 1, cb = (tid & 1) * 64;    // row = s
        float dec = __expf(sAcs[CHUNK - 1] - sAcs[row]);
#pragma unroll
        for (int i = 0; i < 64; ++i) {
            float v = (float)sB[row * D_STATE + cb + i] * dec;
            sBT[(size_t)(cb + i) * CHUNK + row] = bfc(v);
        }
    }
    __syncthreads();

    // ---- Phase 2: Y_diag[t][d] = sum_s GL[t][s] * xdt[s][d] -> yacc
    {
        const int tm = wave * 16;
#pragma unroll
        for (int d4 = 0; d4 < 4; ++d4) {
            v8f acc = {};
#pragma unroll
            for (int ks = 0; ks < 4; ++ks) {
                v16bf a = load_a(sGL, CHUNK, tm, ks * 32, lane);
                v16bf bb = load_b(sXT, CHUNK, d4 * 16, ks * 32, lane);
                acc = wmma_bf16(a, bb, acc);
            }
            const int nl = lane & 15, mh = (lane >> 4) * 8;
            float* o = yacc + (tokBase + tm + mh) * D_INNER + (size_t)h * HEAD_D + d4 * 16 + nl;
#pragma unroll
            for (int r = 0; r < 8; ++r) o[(size_t)r * D_INNER] = acc[r];
        }
    }

    // ---- Phase 3: states[d][n] = sum_s xdt[s][d] * decayB[s][n]
    {
        const int dm = (wave & 3) * 16;
        const int nb = (wave >> 2) * 64;
        float* sOut = states + ((((size_t)b * NCHUNK + c) * N_HEADS + h) * HEAD_D) * D_STATE;
#pragma unroll
        for (int n4 = 0; n4 < 4; ++n4) {
            v8f acc = {};
#pragma unroll
            for (int ks = 0; ks < 4; ++ks) {
                v16bf a = load_a(sXT, CHUNK, dm, ks * 32, lane);
                v16bf bb = load_b(sBT, CHUNK, nb + n4 * 16, ks * 32, lane);
                acc = wmma_bf16(a, bb, acc);
            }
            const int nl = lane & 15, mh = (lane >> 4) * 8;
#pragma unroll
            for (int r = 0; r < 8; ++r)
                sOut[(size_t)(dm + mh + r) * D_STATE + nb + n4 * 16 + nl] = acc[r];
        }
    }
}

// ================= 6) inter-chunk state scan (in place: states <- prev)
__global__ __launch_bounds__(256) void k_scan(const float* __restrict__ acs,
                                              float* __restrict__ states) {
    const int h = blockIdx.x, b = blockIdx.y;
    const int tid = threadIdx.x;
    const size_t hb = (size_t)b * N_HEADS + h;
    float carry[32];
#pragma unroll
    for (int i = 0; i < 32; ++i) carry[i] = 0.f;
    for (int c = 0; c < NCHUNK; ++c) {
        float dec = __expf(acs[(hb * NCHUNK + c) * CHUNK + (CHUNK - 1)]);
        float* p = states + (((size_t)b * NCHUNK + c) * N_HEADS + h) * (size_t)(HEAD_D * D_STATE);
#pragma unroll
        for (int i = 0; i < 32; ++i) {
            size_t e = (size_t)tid + (size_t)i * 256;
            float s_old = p[e];
            p[e] = carry[i];                      // prev[c]
            carry[i] = dec * carry[i] + s_old;    // prev[c+1]
        }
    }
}

// ================= 7) Y_off = exp(acs[t]) * (C @ prev^T), + D-residual, SiLU gate
__global__ __launch_bounds__(256) void k_yoff(const float* __restrict__ xbc,
                                              const float* __restrict__ proj,
                                              const float* __restrict__ prevS,
                                              const float* __restrict__ acs,
                                              const float* __restrict__ Dp,
                                              float* __restrict__ yacc) {
    const int h = blockIdx.x, c = blockIdx.y, b = blockIdx.z;
    __shared__ __bf16 sC[CHUNK * D_STATE];    // [t][n]
    __shared__ __bf16 sP[HEAD_D * D_STATE];   // [d][n]
    __shared__ float  sAcs[CHUNK];
    const int tid = threadIdx.x, lane = tid & 31, wave = tid >> 5;
    const size_t tokBase = (size_t)b * L_SZ + (size_t)c * CHUNK;
    if (tid < CHUNK)
        sAcs[tid] = acs[(((size_t)b * N_HEADS + h) * NCHUNK + c) * CHUNK + tid];
    {
        int row = tid >> 1, cb = (tid & 1) * 64;
        const float* gC = xbc + (tokBase + row) * CONV_DIM + D_INNER + D_STATE + cb;
#pragma unroll
        for (int i = 0; i < 64; i += 4) {
            float4 v = *(const float4*)(gC + i);
            sC[row * D_STATE + cb + i + 0] = bfc(v.x);
            sC[row * D_STATE + cb + i + 1] = bfc(v.y);
            sC[row * D_STATE + cb + i + 2] = bfc(v.z);
            sC[row * D_STATE + cb + i + 3] = bfc(v.w);
        }
    }
    {
        int row = tid >> 2, cb = (tid & 3) * 32;  // row = d
        const float* gp = prevS + ((((size_t)b * NCHUNK + c) * N_HEADS + h) * HEAD_D + row)
                                      * D_STATE + cb;
#pragma unroll
        for (int i = 0; i < 32; i += 4) {
            float4 v = *(const float4*)(gp + i);
            sP[row * D_STATE + cb + i + 0] = bfc(v.x);
            sP[row * D_STATE + cb + i + 1] = bfc(v.y);
            sP[row * D_STATE + cb + i + 2] = bfc(v.z);
            sP[row * D_STATE + cb + i + 3] = bfc(v.w);
        }
    }
    __syncthreads();
    const float Dh = Dp[h];
    const int tm = wave * 16;
#pragma unroll
    for (int d4 = 0; d4 < 4; ++d4) {
        v8f acc = {};
#pragma unroll
        for (int ks = 0; ks < 4; ++ks) {
            v16bf a = load_a(sC, D_STATE, tm, ks * 32, lane);
            v16bf bb = load_b(sP, D_STATE, d4 * 16, ks * 32, lane);
            acc = wmma_bf16(a, bb, acc);
        }
        const int nl = lane & 15, mh = (lane >> 4) * 8;
        const int d = d4 * 16 + nl;
#pragma unroll
        for (int r = 0; r < 8; ++r) {
            int t = tm + mh + r;
            size_t tok = tokBase + t;
            size_t ch = (size_t)h * HEAD_D + d;
            float xr = xbc[tok * CONV_DIM + ch];
            float g = proj[tok * PROJ_DIM + ch];
            float y = yacc[tok * D_INNER + ch] + __expf(sAcs[t]) * acc[r] + Dh * xr;
            yacc[tok * D_INNER + ch] = y * siluf(g);
        }
    }
}

// ================= 8) gated RMSNorm -> bf16
__global__ __launch_bounds__(256) void k_norm(const float* __restrict__ yf,
                                              const float* __restrict__ nw,
                                              __bf16* __restrict__ out) {
    const size_t tok = blockIdx.x;
    __shared__ float red[256];
    const float* row = yf + tok * D_INNER;
    float s = 0.f;
    for (int i = threadIdx.x; i < D_INNER; i += 256) { float v = row[i]; s += v * v; }
    red[threadIdx.x] = s;
    __syncthreads();
    for (int o = 128; o > 0; o >>= 1) {
        if ((int)threadIdx.x < o) red[threadIdx.x] += red[threadIdx.x + o];
        __syncthreads();
    }
    const float scale = rsqrtf(red[0] / (float)D_INNER + EPSF);
    __bf16* o = out + tok * D_INNER;
    for (int i = threadIdx.x; i < D_INNER; i += 256) o[i] = bfc(row[i] * scale * nw[i]);
}

// ================= host launcher =================
extern "C" void kernel_launch(void* const* d_in, const int* in_sizes, int n_in,
                              void* d_out, int out_size, void* d_ws, size_t ws_size,
                              hipStream_t stream) {
    const float* hid     = (const float*)d_in[0];
    const float* in_w    = (const float*)d_in[1];
    const float* conv_w  = (const float*)d_in[2];
    const float* conv_b  = (const float*)d_in[3];
    const float* dt_bias = (const float*)d_in[4];
    const float* A_log   = (const float*)d_in[5];
    const float* Dp      = (const float*)d_in[6];
    const float* norm_w  = (const float*)d_in[7];
    const float* out_w   = (const float*)d_in[8];

    char* ws = (char*)d_ws;
    float* proj    = (float*)ws;  ws += (size_t)TOK * PROJ_DIM * 4;
    float* xbc     = (float*)ws;  ws += (size_t)TOK * CONV_DIM * 4;
    float* dtb     = (float*)ws;  ws += (size_t)TOK * N_HEADS * 4;
    float* acs     = (float*)ws;  ws += (size_t)B_SZ * N_HEADS * NCHUNK * CHUNK * 4;
    float* states  = (float*)ws;  ws += (size_t)B_SZ * NCHUNK * N_HEADS * HEAD_D * D_STATE * 4;
    float* yacc    = (float*)ws;  ws += (size_t)TOK * D_INNER * 4;
    __bf16* normed = (__bf16*)ws; ws += (size_t)TOK * D_INNER * 2;
    __bf16* hidb   = (__bf16*)ws; ws += (size_t)TOK * HDIM * 2;
    __bf16* wib    = (__bf16*)ws; ws += (size_t)PROJ_DIM * HDIM * 2;
    __bf16* wob    = (__bf16*)ws; ws += (size_t)HDIM * D_INNER * 2;

    const size_t nHid = (size_t)TOK * HDIM;
    const size_t nWi  = (size_t)PROJ_DIM * HDIM;
    const size_t nWo  = (size_t)HDIM * D_INNER;
    k_f2bf<<<(unsigned)((nHid / 4 + 255) / 256), 256, 0, stream>>>(hid, hidb, nHid);
    k_f2bf<<<(unsigned)((nWi  / 4 + 255) / 256), 256, 0, stream>>>(in_w, wib, nWi);
    k_f2bf<<<(unsigned)((nWo  / 4 + 255) / 256), 256, 0, stream>>>(out_w, wob, nWo);

    k_gemm_bf16<<<dim3(PROJ_DIM / BN, TOK / BM), 256, 0, stream>>>(hidb, wib, proj,
                                                                   TOK, PROJ_DIM, HDIM);
    k_conv<<<(unsigned)(((size_t)TOK * CONV_DIM + 255) / 256), 256, 0, stream>>>(proj, conv_w,
                                                                                 conv_b, xbc);
    k_dt<<<(TOK * N_HEADS + 255) / 256, 256, 0, stream>>>(proj, dt_bias, dtb);
    k_acs<<<(B_SZ * N_HEADS * NCHUNK + 255) / 256, 256, 0, stream>>>(dtb, A_log, acs);
    k_chunk<<<dim3(N_HEADS, NCHUNK, B_SZ), 256, 0, stream>>>(xbc, dtb, acs, states, yacc);
    k_scan<<<dim3(N_HEADS, B_SZ), 256, 0, stream>>>(acs, states);
    k_yoff<<<dim3(N_HEADS, NCHUNK, B_SZ), 256, 0, stream>>>(xbc, proj, states, acs, Dp, yacc);
    k_norm<<<TOK, 256, 0, stream>>>(yacc, norm_w, normed);
    k_gemm_bf16<<<dim3(HDIM / BN, TOK / BM), 256, 0, stream>>>(normed, wob, (float*)d_out,
                                                               TOK, HDIM, D_INNER);
}
